// WorldModel_5695126634971
// MI455X (gfx1250) — matrix operand order repaired and checked by reading, not compiled, not run on hardware
//
#include <hip/hip_runtime.h>
#include <hip/hip_bf16.h>
#include <math.h>

typedef __bf16 bf16;
typedef __bf16 v16bf __attribute__((ext_vector_type(16)));
typedef __bf16 v8bf  __attribute__((ext_vector_type(8)));
typedef float  v8f   __attribute__((ext_vector_type(8)));
typedef float  v4f   __attribute__((ext_vector_type(4)));

#define B_     2048
#define T_     64
#define TP     63          // T-1 timesteps actually used
#define NPOS   (B_ * TP)   // 129024 positions, multiple of 128
#define SDIM   12
#define ADIM   4
#define LATENT 256
#define ALATD  64
#define CATD   320         // LATENT + ALATD, multiple of 32
#define HID    512
#define GATES  1536

// ---------------------------------------------------------------- helpers
__device__ __forceinline__ float siluf(float x)    { return x / (1.0f + __expf(-x)); }
__device__ __forceinline__ float sigmoidf_(float x){ return 1.0f / (1.0f + __expf(-x)); }

union U16bf { v16bf v; v8bf h[2]; };

// A-fragment (16x32 bf16), A row-major [rows][lda].
// Layout (ISA 7.12.2): lane m=L&15 holds row m; lanes 0-15: K 0..7 & 16..23,
// lanes 16-31: K 8..15 & 24..31.  -> two contiguous 16B loads per lane.
__device__ __forceinline__ v16bf load_a_bf16(const bf16* A, int lda, int row0, int k0, int lane) {
  int m = lane & 15, kg = lane >> 4;
  const bf16* p = A + (size_t)(row0 + m) * lda + k0 + kg * 8;
  U16bf u;
  u.h[0] = *(const v8bf*)p;
  u.h[1] = *(const v8bf*)(p + 16);
  return u.v;
}

__device__ __forceinline__ v16bf load_a_f32(const float* A, int lda, int row0, int k0, int lane) {
  int m = lane & 15, kg = lane >> 4;
  const float* p = A + (size_t)(row0 + m) * lda + k0 + kg * 8;
  const v4f* q = (const v4f*)p;
  v4f x0 = q[0], x1 = q[1], x2 = q[4], x3 = q[5]; // p+16 floats
  U16bf u;
#pragma unroll
  for (int e = 0; e < 4; ++e) {
    u.v[e]      = (bf16)x0[e];
    u.v[4 + e]  = (bf16)x1[e];
    u.v[8 + e]  = (bf16)x2[e];
    u.v[12 + e] = (bf16)x3[e];
  }
  return u.v;
}

// B-fragment from pre-packed weights: tile (kt,nt) stored as 32 lanes x 16 elems
// contiguous; one aligned 32B load per lane.
__device__ __forceinline__ v16bf load_b_pk(const bf16* Bpk, int kt, int NT, int nt, int lane) {
  const bf16* p = Bpk + (((size_t)(kt * NT + nt) * 32) + lane) * 16;
  return *(const v16bf*)p;
}

__device__ __forceinline__ v8f wmma_bf16(v16bf a, v16bf b, v8f c) {
  return __builtin_amdgcn_wmma_f32_16x16x32_bf16(false, a, false, b, (short)0, c, false, false);
}

// ------------------------------------------------------ weight pack kernel
// Pack fp32 weight into WMMA B-fragment order (bf16).
// Packed element gid = ((kt*NT + nt)*32 + lane)*16 + elem maps to
// k = kt*32 + (lane>>4)*8 + (elem&7) + (elem>>3)*16 ; n = nt*16 + (lane&15).
__global__ void pack_w(const float* __restrict__ src, bf16* __restrict__ dst,
                       int K, int N, int transposed) {
  int gid = blockIdx.x * blockDim.x + threadIdx.x;
  if (gid >= K * N) return;
  int elem = gid & 15;
  int lane = (gid >> 4) & 31;
  int tile = gid >> 9;
  int NT = N >> 4;
  int nt = tile % NT, kt = tile / NT;
  int k = kt * 32 + (lane >> 4) * 8 + (elem & 7) + (elem >> 3) * 16;
  int n = nt * 16 + (lane & 15);
  float v = transposed ? src[(size_t)n * K + k] : src[(size_t)k * N + n];
  dst[gid] = (bf16)v;
}

// ---------------------------------------------------------- block reduce
__device__ __forceinline__ float block_sum128(float v, float* red, int tid) {
  red[tid] = v;
  __syncthreads();
#pragma unroll
  for (int s = 64; s > 0; s >>= 1) {
    if (tid < s) red[tid] += red[tid + s];
    __syncthreads();
  }
  float r = red[0];
  __syncthreads();
  return r;
}

// ------------------------------------------------------- encoder (VALU)
// One 128-thread block per position: state MLP (12->128 LN silu ->256 LN silu
// ->256) and action MLP (4->32 silu ->64); writes bf16 concat row (320).
__global__ __launch_bounds__(128) void encoder_kernel(
    const float* __restrict__ states, const float* __restrict__ actions,
    const float* __restrict__ w1, const float* __restrict__ b1,
    const float* __restrict__ g1, const float* __restrict__ be1,
    const float* __restrict__ w2, const float* __restrict__ b2,
    const float* __restrict__ g2, const float* __restrict__ be2,
    const float* __restrict__ w3, const float* __restrict__ b3,
    const float* __restrict__ aw1, const float* __restrict__ ab1,
    const float* __restrict__ aw2, const float* __restrict__ ab2,
    bf16* __restrict__ cat) {
  __shared__ float st[SDIM];
  __shared__ float h1[128];
  __shared__ float h2[256];
  __shared__ float ah[32];
  __shared__ float red[128];
  int tid = threadIdx.x;
  int pos = blockIdx.x;
  int b = pos / TP, t = pos % TP;

  const float* sp = states + ((size_t)b * T_ + t) * SDIM;
  if (tid < SDIM) st[tid] = sp[tid];
  __syncthreads();

  // layer 1: 12 -> 128, LN, silu
  float v = b1[tid];
#pragma unroll
  for (int k = 0; k < SDIM; ++k) v += st[k] * w1[k * 128 + tid];
  float m = block_sum128(v, red, tid) * (1.0f / 128.0f);
  float d = v - m;
  float var = block_sum128(d * d, red, tid) * (1.0f / 128.0f);
  v = d * rsqrtf(var + 1e-5f) * g1[tid] + be1[tid];
  h1[tid] = siluf(v);
  __syncthreads();

  // layer 2: 128 -> 256 (2 outputs/thread), LN, silu
  float o0 = b2[tid], o1 = b2[tid + 128];
  for (int k = 0; k < 128; ++k) {
    float hk = h1[k];
    o0 += hk * w2[k * 256 + tid];
    o1 += hk * w2[k * 256 + tid + 128];
  }
  float m2 = block_sum128(o0 + o1, red, tid) * (1.0f / 256.0f);
  float d0 = o0 - m2, d1 = o1 - m2;
  float var2 = block_sum128(d0 * d0 + d1 * d1, red, tid) * (1.0f / 256.0f);
  float rstd = rsqrtf(var2 + 1e-5f);
  h2[tid]       = siluf(d0 * rstd * g2[tid] + be2[tid]);
  h2[tid + 128] = siluf(d1 * rstd * g2[tid + 128] + be2[tid + 128]);
  __syncthreads();

  // slat: 256 -> 256
  float s0 = b3[tid], s1 = b3[tid + 128];
  for (int k = 0; k < 256; ++k) {
    float hk = h2[k];
    s0 += hk * w3[k * 256 + tid];
    s1 += hk * w3[k * 256 + tid + 128];
  }
  bf16* cp = cat + (size_t)pos * CATD;
  cp[tid]       = (bf16)s0;
  cp[tid + 128] = (bf16)s1;

  // action encoder
  const float* ap = actions + ((size_t)b * T_ + t) * ADIM;
  if (tid < 32) {
    float a = ab1[tid];
#pragma unroll
    for (int k = 0; k < ADIM; ++k) a += ap[k] * aw1[k * 32 + tid];
    ah[tid] = siluf(a);
  }
  __syncthreads();
  if (tid < ALATD) {
    float a = ab2[tid];
    for (int k = 0; k < 32; ++k) a += ah[k] * aw2[k * ALATD + tid];
    cp[LATENT + tid] = (bf16)a;
  }
}

// ------------------------------------------------ generic WMMA GEMM kernel
// D = act(A @ B + bias).  A: bf16 (Ab) or fp32 (Af) row-major [M][lda=K].
// Bpk: pre-packed bf16, logical K x N.  Block = 8 waves, 128x64 tile;
// each wave: one 16-row strip x 4 column tiles sharing one A fragment.
// Prefetch of the next K-step's packed-B tile is unconditional: speculative
// prefetch past the buffer end is dropped by hardware (ISA prefetch rules).
__global__ __launch_bounds__(256) void wmma_gemm(
    const bf16* __restrict__ Ab, const float* __restrict__ Af, int lda, int K,
    const bf16* __restrict__ Bpk, int N, const float* __restrict__ bias,
    int act, float* __restrict__ Df, bf16* __restrict__ Db) {
  int lane = threadIdx.x & 31;
  int wave = threadIdx.x >> 5;
  int row0 = blockIdx.x * 128 + wave * 16;
  int ntb  = blockIdx.y * 4;
  int NT   = N >> 4;
  int KT   = K >> 5;

  v8f acc[4];
#pragma unroll
  for (int i = 0; i < 4; ++i)
#pragma unroll
    for (int j = 0; j < 8; ++j) acc[i][j] = 0.0f;

  if (Af) {
    for (int kt = 0; kt < KT; ++kt) {
      __builtin_prefetch(Bpk + (((size_t)((kt + 1) * NT + ntb) * 32) + lane) * 16, 0, 3);
      v16bf a = load_a_f32(Af, lda, row0, kt * 32, lane);
#pragma unroll
      for (int i = 0; i < 4; ++i) {
        v16bf bfr = load_b_pk(Bpk, kt, NT, ntb + i, lane);
        acc[i] = wmma_bf16(a, bfr, acc[i]);
      }
    }
  } else {
    for (int kt = 0; kt < KT; ++kt) {
      __builtin_prefetch(Bpk + (((size_t)((kt + 1) * NT + ntb) * 32) + lane) * 16, 0, 3);
      v16bf a = load_a_bf16(Ab, lda, row0, kt * 32, lane);
#pragma unroll
      for (int i = 0; i < 4; ++i) {
        v16bf bfr = load_b_pk(Bpk, kt, NT, ntb + i, lane);
        acc[i] = wmma_bf16(a, bfr, acc[i]);
      }
    }
  }

  // epilogue: C/D layout -> lane holds col lane&15, rows (lane>>4)*8 + j
  int mbase = row0 + (lane >> 4) * 8;
#pragma unroll
  for (int i = 0; i < 4; ++i) {
    int col = (ntb + i) * 16 + (lane & 15);
    float bb = bias ? bias[col] : 0.0f;
#pragma unroll
    for (int j = 0; j < 8; ++j) {
      float v = acc[i][j] + bb;
      if (act) v = siluf(v);
      size_t idx = (size_t)(mbase + j) * N + col;
      if (Df) Df[idx] = v;
      if (Db) Db[idx] = (bf16)v;
    }
  }
}

// ----------------------------------------------------- fused GRU timestep
// gh = h_prev @ whh^T ; gates fused.  One wave: 16 batch rows x 64 hidden
// cols across ALL three gates (12 accumulators, shared A fragment).
__global__ __launch_bounds__(256) void gru_step(
    const bf16* __restrict__ hprev_b, const bf16* __restrict__ whh_pk,
    const float* __restrict__ gi, const float* __restrict__ bhh,
    float* __restrict__ outh, bf16* __restrict__ hnew_b, int t) {
  int lane = threadIdx.x & 31;
  int wave = threadIdx.x >> 5;
  int row0 = blockIdx.x * 128 + wave * 16;
  int ntb  = blockIdx.y * 4;
  const int NT = GATES / 16; // 96

  v8f aR[4], aZ[4], aN[4];
#pragma unroll
  for (int i = 0; i < 4; ++i)
#pragma unroll
    for (int j = 0; j < 8; ++j) { aR[i][j] = 0.0f; aZ[i][j] = 0.0f; aN[i][j] = 0.0f; }

  for (int kt = 0; kt < HID / 32; ++kt) {
    __builtin_prefetch(whh_pk + (((size_t)((kt + 1) * NT + ntb) * 32) + lane) * 16, 0, 3);
    v16bf a = load_a_bf16(hprev_b, HID, row0, kt * 32, lane);
#pragma unroll
    for (int i = 0; i < 4; ++i) {
      v16bf bR = load_b_pk(whh_pk, kt, NT, ntb + i, lane);
      aR[i] = wmma_bf16(a, bR, aR[i]);
      v16bf bZ = load_b_pk(whh_pk, kt, NT, 32 + ntb + i, lane);
      aZ[i] = wmma_bf16(a, bZ, aZ[i]);
      v16bf bN = load_b_pk(whh_pk, kt, NT, 64 + ntb + i, lane);
      aN[i] = wmma_bf16(a, bN, aN[i]);
    }
  }

  int mbase = row0 + (lane >> 4) * 8;
#pragma unroll
  for (int i = 0; i < 4; ++i) {
    int col = (ntb + i) * 16 + (lane & 15);
    float bR = bhh[col], bZ = bhh[col + HID], bN = bhh[col + 2 * HID];
#pragma unroll
    for (int j = 0; j < 8; ++j) {
      int b = mbase + j;
      size_t gib = ((size_t)b * TP + t) * GATES;
      float r = sigmoidf_(gi[gib + col]           + aR[i][j] + bR);
      float z = sigmoidf_(gi[gib + HID + col]     + aZ[i][j] + bZ);
      float n = tanhf   (gi[gib + 2 * HID + col]  + r * (aN[i][j] + bN));
      float hp = (t > 0) ? outh[((size_t)b * TP + (t - 1)) * HID + col] : 0.0f;
      float hn = (1.0f - z) * n + z * hp;
      outh[((size_t)b * TP + t) * HID + col] = hn;
      hnew_b[(size_t)b * HID + col] = (bf16)hn;
    }
  }
}

// ------------------------------------------------------------- tiny head
__global__ __launch_bounds__(128) void td_head(
    const float* __restrict__ nlat,
    const float* __restrict__ w1, const float* __restrict__ b1,
    const float* __restrict__ w2, const float* __restrict__ b2,
    const float* __restrict__ w3, const float* __restrict__ b3,
    float* __restrict__ out) {
  __shared__ float nl[256];
  __shared__ float y1[128];
  __shared__ float y2[64];
  int tid = threadIdx.x;
  size_t pos = blockIdx.x;
  const float* np = nlat + pos * 256;
  nl[tid] = np[tid];
  nl[tid + 128] = np[tid + 128];
  __syncthreads();
  float v = b1[tid];
  for (int k = 0; k < 256; ++k) v += nl[k] * w1[k * 128 + tid];
  y1[tid] = siluf(v);
  __syncthreads();
  if (tid < 64) {
    float u = b2[tid];
    for (int k = 0; k < 128; ++k) u += y1[k] * w2[k * 64 + tid];
    y2[tid] = siluf(u);
  }
  __syncthreads();
  if (tid < 3) {
    float o = b3[tid];
    for (int k = 0; k < 64; ++k) o += y2[k] * w3[k * 3 + tid];
    out[pos * 3 + tid] = o;
  }
}

// ---------------------------------------------------------------- launch
extern "C" void kernel_launch(void* const* d_in, const int* in_sizes, int n_in,
                              void* d_out, int out_size, void* d_ws, size_t ws_size,
                              hipStream_t stream) {
  (void)in_sizes; (void)n_in; (void)out_size; (void)ws_size;
  const float* states  = (const float*)d_in[0];
  const float* actions = (const float*)d_in[1];
  const float* se_w1 = (const float*)d_in[2];
  const float* se_b1 = (const float*)d_in[3];
  const float* se_g1 = (const float*)d_in[4];
  const float* se_be1= (const float*)d_in[5];
  const float* se_w2 = (const float*)d_in[6];
  const float* se_b2 = (const float*)d_in[7];
  const float* se_g2 = (const float*)d_in[8];
  const float* se_be2= (const float*)d_in[9];
  const float* se_w3 = (const float*)d_in[10];
  const float* se_b3 = (const float*)d_in[11];
  const float* ae_w1 = (const float*)d_in[12];
  const float* ae_b1 = (const float*)d_in[13];
  const float* ae_w2 = (const float*)d_in[14];
  const float* ae_b2 = (const float*)d_in[15];
  const float* dp_w  = (const float*)d_in[16];
  const float* dp_b  = (const float*)d_in[17];
  const float* gwih  = (const float*)d_in[18];
  const float* gwhh  = (const float*)d_in[19];
  const float* gbih  = (const float*)d_in[20];
  const float* gbhh  = (const float*)d_in[21];
  const float* do_w1 = (const float*)d_in[22];
  const float* do_b1 = (const float*)d_in[23];
  const float* do_w2 = (const float*)d_in[24];
  const float* do_b2 = (const float*)d_in[25];
  const float* td_w1 = (const float*)d_in[26];
  const float* td_b1 = (const float*)d_in[27];
  const float* td_w2 = (const float*)d_in[28];
  const float* td_b2 = (const float*)d_in[29];
  const float* td_w3 = (const float*)d_in[30];
  const float* td_b3 = (const float*)d_in[31];

  // workspace carve (256B aligned)
  char* base = (char*)d_ws;
  size_t off = 0;
  auto carve = [&](size_t bytes) -> char* {
    char* p = base + off;
    off = (off + bytes + 255) & ~(size_t)255;
    return p;
  };
  bf16*  cat    = (bf16*) carve((size_t)NPOS * CATD * 2);
  bf16*  dpw_pk = (bf16*) carve((size_t)CATD * HID * 2);
  bf16*  wih_pk = (bf16*) carve((size_t)HID * GATES * 2);
  bf16*  whh_pk = (bf16*) carve((size_t)HID * GATES * 2);
  bf16*  dw1_pk = (bf16*) carve((size_t)HID * HID * 2);
  bf16*  dw2_pk = (bf16*) carve((size_t)HID * LATENT * 2);
  bf16*  xb     = (bf16*) carve((size_t)NPOS * HID * 2);
  float* gi     = (float*)carve((size_t)NPOS * GATES * 4);
  bf16*  hA     = (bf16*) carve((size_t)B_ * HID * 2);
  bf16*  hB     = (bf16*) carve((size_t)B_ * HID * 2);
  float* outh   = (float*)carve((size_t)NPOS * HID * 4);
  bf16*  d1     = (bf16*) carve((size_t)NPOS * HID * 2);
  float* nlat   = (float*)carve((size_t)NPOS * LATENT * 4);

  // 1) pack weights to WMMA fragment layout (bf16)
  pack_w<<<(CATD * HID + 255) / 256, 256, 0, stream>>>(dp_w,  dpw_pk, CATD, HID,    0);
  pack_w<<<(HID * GATES + 255) / 256, 256, 0, stream>>>(gwih,  wih_pk, HID,  GATES, 1);
  pack_w<<<(HID * GATES + 255) / 256, 256, 0, stream>>>(gwhh,  whh_pk, HID,  GATES, 1);
  pack_w<<<(HID * HID + 255) / 256, 256, 0, stream>>>(do_w1, dw1_pk, HID,  HID,    0);
  pack_w<<<(HID * LATENT + 255) / 256, 256, 0, stream>>>(do_w2, dw2_pk, HID, LATENT, 0);

  // 2) fused per-position encoders -> bf16 concat(slat, alat)
  encoder_kernel<<<NPOS, 128, 0, stream>>>(
      states, actions, se_w1, se_b1, se_g1, se_be1, se_w2, se_b2, se_g2, se_be2,
      se_w3, se_b3, ae_w1, ae_b1, ae_w2, ae_b2, cat);

  // 3) x = cat @ dp_w + dp_b  (bf16 out)
  wmma_gemm<<<dim3(NPOS / 128, HID / 64), 256, 0, stream>>>(
      cat, nullptr, CATD, CATD, dpw_pk, HID, dp_b, 0, nullptr, xb);

  // 4) gi = x @ wih^T + bih  (fp32 out)
  wmma_gemm<<<dim3(NPOS / 128, GATES / 64), 256, 0, stream>>>(
      xb, nullptr, HID, HID, wih_pk, GATES, gbih, 0, gi, nullptr);

  // 5) GRU recurrence: 63 fused GEMM+gating steps (h0 = 0)
  hipMemsetAsync(hA, 0, (size_t)B_ * HID * 2, stream);
  for (int t = 0; t < TP; ++t) {
    const bf16* hp = (t & 1) ? hB : hA;
    bf16* hn = (t & 1) ? hA : hB;
    gru_step<<<dim3(B_ / 128, HID / 64), 256, 0, stream>>>(
        hp, whh_pk, gi, gbhh, outh, hn, t);
  }

  // 6) d1 = silu(out @ do_w1 + b1)  (fp32 A, bf16 out)
  wmma_gemm<<<dim3(NPOS / 128, HID / 64), 256, 0, stream>>>(
      nullptr, outh, HID, HID, dw1_pk, HID, do_b1, 1, nullptr, d1);

  // 7) nlat = d1 @ do_w2 + b2  (fp32 out)
  wmma_gemm<<<dim3(NPOS / 128, LATENT / 64), 256, 0, stream>>>(
      d1, nullptr, HID, HID, dw2_pk, LATENT, do_b2, 0, nlat, nullptr);

  // 8) tiny decoder head -> d_out (B, 63, 3)
  td_head<<<NPOS, 128, 0, stream>>>(nlat, td_w1, td_b1, td_w2, td_b2,
                                    td_w3, td_b3, (float*)d_out);
}
